// LSTM_Module_88218628260031
// MI455X (gfx1250) — compile-verified, matrix operand order checked by reading, not run on hardware
//
#include <hip/hip_runtime.h>
#include <hip/hip_bf16.h>

typedef __bf16        v16bf  __attribute__((ext_vector_type(16)));
typedef float         v8f    __attribute__((ext_vector_type(8)));
typedef unsigned int  u32x4  __attribute__((ext_vector_type(4)));
typedef unsigned short u16;

#define LSTM_H   1024
#define LSTM_T   2048
#define LSTM_IN  5
#define NBLK     32      // persistent blocks (must be co-resident)
#define BTHREADS 256     // 8 waves of 32

union ABv { v16bf v; u32x4 q[2]; };

__device__ __forceinline__ u16 f2bf(float f) {
    union { float f; unsigned u; } x; x.f = f;
    unsigned r = x.u + 0x7FFFu + ((x.u >> 16) & 1u);   // round-to-nearest-even
    return (u16)(r >> 16);
}

// ---------------------------------------------------------------------------
// init: zero double-buffered h_bf16 and grid-barrier state (every call, so
// graph replays are deterministic).
// ---------------------------------------------------------------------------
__global__ void init_state(u16* h_bf, unsigned* bar) {
    int t = blockIdx.x * blockDim.x + threadIdx.x;
    if (t < 2 * LSTM_H) h_bf[t] = 0;
    if (t == 0) { bar[0] = 0u; bar[1] = 0u; }
}

// ---------------------------------------------------------------------------
// pack w_hh (fp32 [4096][1024] row-major) -> bf16 WMMA B-tile layout.
// Tile (nt,kt): 16 rows (N) x 32 K. Per lane l, 16 contiguous bf16 values:
//   lane<16  : column n=l,    K = base+{0..7}, base+16+{0..7}, base=0
//   lane>=16 : column n=l-16, same with base=8
// packed[tile*512 + lane*16 + i]
// ---------------------------------------------------------------------------
__global__ void pack_whh(const float* __restrict__ w_hh, u16* __restrict__ wpack) {
    int idx = blockIdx.x * blockDim.x + threadIdx.x;      // 0 .. 4M-1
    if (idx >= 4096 * 1024) return;
    int tt   = idx >> 9;            // tile id (nt*32 + kt)
    int lane = (idx >> 4) & 31;
    int i    = idx & 15;
    int nt   = tt >> 5;             // 0..255
    int kt   = tt & 31;             // 0..31
    int row  = nt * 16 + (lane & 15);
    int base = (lane & 16) ? 8 : 0;
    int koff = base + i + ((i & 8) ? 8 : 0);              // {base..base+7, base+16..base+23}
    int col  = kt * 32 + koff;
    wpack[idx] = f2bf(w_hh[row * 1024 + col]);
}

// ---------------------------------------------------------------------------
// grid-wide barrier (sense-reversal via phase counter).
// ---------------------------------------------------------------------------
__device__ __forceinline__ void grid_sync(unsigned* cnt, unsigned* ph,
                                          unsigned nb, unsigned& lp) {
    __syncthreads();
    if (threadIdx.x == 0) {
        __threadfence();
        unsigned prev = atomicAdd(cnt, 1u);
        if (prev == nb - 1u) {
            atomicExch(cnt, 0u);
            __threadfence();
            atomicAdd(ph, 1u);
        } else {
            volatile unsigned* vph = (volatile unsigned*)ph;
            while (*vph <= lp) { __builtin_amdgcn_s_sleep(2); }
        }
        __threadfence();
    }
    lp++;
    __syncthreads();
}

// ---------------------------------------------------------------------------
// Persistent LSTM recurrence with VGPR-resident weights.
// Block b owns hidden units [32b, 32b+32) (8 n-tiles -> 8 waves). Each wave
// preloads its 32 B-tiles (32 KB -> 256 VGPRs) ONCE; the 2048-step loop then
// touches memory only for the 2 KB h broadcast per step. A-fragment DS reads
// are pinned 4 tiles ahead of the WMMA chain via sched_group_barrier so the
// waits become partial instead of s_wait_dscnt 0.
// ---------------------------------------------------------------------------
__global__ void __launch_bounds__(BTHREADS, 1)
lstm_persistent(const float* __restrict__ x,
                const float* __restrict__ w_ih,
                const float* __restrict__ b_ih,
                const float* __restrict__ b_hh,
                const u16*   __restrict__ wpack,
                u16*         __restrict__ h_bf,     // [2][1024] double buffer
                float*       __restrict__ h_f32,    // [1024]
                unsigned*                 bar) {    // {count, phase}
    __shared__ u16   h_lds[LSTM_H];                  // 2 KB
    __shared__ float g_lds[128];                     // block's 128 gate pre-acts
    __shared__ float wih_l[128 * LSTM_IN];
    __shared__ float bsum_l[128];
    __shared__ float x_lds[LSTM_T * LSTM_IN];        // 40 KB: whole input sequence

    const int b    = blockIdx.x;
    const int tid  = threadIdx.x;
    const int wave = tid >> 5;
    const int lane = tid & 31;

    // per-block input-projection params: local row lr -> global row
    for (int lr = tid; lr < 128; lr += BTHREADS) {
        int j = (lr >> 5) * 1024 + b * 32 + (lr & 31);
        #pragma unroll
        for (int i = 0; i < LSTM_IN; ++i) wih_l[lr * LSTM_IN + i] = w_ih[j * LSTM_IN + i];
        bsum_l[lr] = b_ih[j] + b_hh[j];
    }
    // stage the full input sequence into LDS
    for (int k = tid; k < LSTM_T * LSTM_IN; k += BTHREADS) x_lds[k] = x[k];

    // ---- hoist this wave's 32 B-tiles into registers (32 KB/wave) ----
    const int  ntile   = (wave >> 1) * 64 + b * 2 + (wave & 1);
    const u16* wp_base = wpack + (size_t)ntile * 32 * 512;
    ABv bw[32];
    #pragma unroll
    for (int kt = 0; kt < 32; ++kt) {
        const u32x4* bp = (const u32x4*)(wp_base + (size_t)kt * 512 + lane * 16);
        bw[kt].q[0] = bp[0];
        bw[kt].q[1] = bp[1];
    }

    float    c_st  = 0.0f;                // cell state, one unit per thread (tid<32)
    unsigned lp    = 0;
    const int      abase    = (lane & 16) ? 8 : 0;
    const unsigned lds_base = (unsigned)(size_t)&h_lds[0];

    auto loadA = [&](int kt) {
        ABv a;
        a.q[0] = *(const u32x4*)&h_lds[kt * 32 + abase];        // K base..base+7
        a.q[1] = *(const u32x4*)&h_lds[kt * 32 + 16 + abase];   // K base+16..base+23
        return a;
    };

    __syncthreads();

    for (int t = 0; t < LSTM_T; ++t) {
        // stage h (2 KB bf16) directly into LDS via the async data path
        const u16* hin = h_bf + (size_t)(t & 1) * LSTM_H;
        if (tid < 128) {
            unsigned           loff = lds_base + (unsigned)tid * 16u;
            unsigned long long ga   = (unsigned long long)hin + (unsigned long long)tid * 16u;
            asm volatile("global_load_async_to_lds_b128 %0, %1, off"
                         :: "v"(loff), "v"(ga) : "memory");
        }
        asm volatile("s_wait_asynccnt 0x0" ::: "memory");
        __syncthreads();

        // g[rows] = W_hh[rows,:] @ h : 32 WMMAs, 2 acc chains, 4-deep A pipeline
        ABv ab[4];
        #pragma unroll
        for (int p = 0; p < 4; ++p) ab[p] = loadA(p);
        v8f acc0 = {};
        v8f acc1 = {};
        #pragma unroll
        for (int kt = 0; kt < 32; ++kt) {
            ABv a = ab[kt & 3];
            if (kt + 4 < 32) ab[kt & 3] = loadA(kt + 4);
            if (kt & 1)
                acc1 = __builtin_amdgcn_wmma_f32_16x16x32_bf16(
                    false, a.v, false, bw[kt].v, (short)0, acc1, false, false);
            else
                acc0 = __builtin_amdgcn_wmma_f32_16x16x32_bf16(
                    false, a.v, false, bw[kt].v, (short)0, acc0, false, false);
        }
#if __has_builtin(__builtin_amdgcn_sched_group_barrier)
        // Pin the pipeline: 8 DS reads up front (4 tiles), then steady-state
        // [1 WMMA, 2 DS reads] x 28, then drain 4 WMMAs. The 64 A-build DS
        // reads are the only DS reads in this scheduling region.
        __builtin_amdgcn_sched_group_barrier(0x100, 8, 0);     // DS read x8
        #pragma unroll
        for (int i = 0; i < 28; ++i) {
            __builtin_amdgcn_sched_group_barrier(0x008, 1, 0); // WMMA x1
            __builtin_amdgcn_sched_group_barrier(0x100, 2, 0); // DS read x2
        }
        __builtin_amdgcn_sched_group_barrier(0x008, 4, 0);     // WMMA x4 drain
#endif
        // A rows replicated -> every D row identical; lane n holds column n
        if (lane < 16) g_lds[wave * 16 + lane] = acc0[0] + acc1[0];
        __syncthreads();

        // elementwise cell update: thread u (<32) owns unit b*32+u, all 4 gates
        if (tid < 32) {
            const float x0 = x_lds[t * LSTM_IN + 0], x1 = x_lds[t * LSTM_IN + 1],
                        x2 = x_lds[t * LSTM_IN + 2], x3 = x_lds[t * LSTM_IN + 3],
                        x4 = x_lds[t * LSTM_IN + 4];
            float gg[4];
            #pragma unroll
            for (int gt = 0; gt < 4; ++gt) {
                int   lr = gt * 32 + tid;
                float v  = g_lds[lr] + bsum_l[lr];
                v += x0 * wih_l[lr * 5 + 0] + x1 * wih_l[lr * 5 + 1] +
                     x2 * wih_l[lr * 5 + 2] + x3 * wih_l[lr * 5 + 3] +
                     x4 * wih_l[lr * 5 + 4];
                gg[gt] = v;
            }
            const float ig = 1.0f / (1.0f + __expf(-gg[0]));
            const float fg = 1.0f / (1.0f + __expf(-gg[1]));
            const float gc = tanhf(gg[2]);
            const float og = 1.0f / (1.0f + __expf(-gg[3]));
            c_st = fg * c_st + ig * gc;
            const float hn = og * tanhf(c_st);
            const int u = b * 32 + tid;
            h_f32[u] = hn;
            h_bf[(size_t)((t + 1) & 1) * LSTM_H + u] = f2bf(hn);
        }
        grid_sync(&bar[0], &bar[1], (unsigned)gridDim.x, lp);
    }
}

// ---------------------------------------------------------------------------
// head: out = log_softmax(tanh(h) @ w_outT + b_out). One block, 9 waves.
// ---------------------------------------------------------------------------
__global__ void head_kernel(const float* __restrict__ h,
                            const float* __restrict__ w_out,
                            const float* __restrict__ b_out,
                            float* __restrict__ out) {
    __shared__ float logits[9];
    const int o = threadIdx.x >> 5;     // 0..8
    const int lane = threadIdx.x & 31;
    float s = 0.0f;
    for (int j = lane; j < LSTM_H; j += 32)
        s += tanhf(h[j]) * w_out[o * LSTM_H + j];
    #pragma unroll
    for (int off = 16; off > 0; off >>= 1) s += __shfl_down(s, off, 32);
    if (lane == 0) logits[o] = s + b_out[o];
    __syncthreads();
    if (threadIdx.x == 0) {
        float m = logits[0];
        for (int i = 1; i < 9; ++i) m = fmaxf(m, logits[i]);
        float se = 0.0f;
        for (int i = 0; i < 9; ++i) se += __expf(logits[i] - m);
        const float lse = m + logf(se);
        for (int i = 0; i < 9; ++i) out[i] = logits[i] - lse;
    }
}

extern "C" void kernel_launch(void* const* d_in, const int* in_sizes, int n_in,
                              void* d_out, int out_size, void* d_ws, size_t ws_size,
                              hipStream_t stream) {
    const float* x     = (const float*)d_in[0];   // [2048,1,5]
    const float* w_ih  = (const float*)d_in[1];   // [4096,5]
    const float* w_hh  = (const float*)d_in[2];   // [4096,1024]
    const float* b_ih  = (const float*)d_in[3];   // [4096]
    const float* b_hh  = (const float*)d_in[4];   // [4096]
    const float* w_out = (const float*)d_in[5];   // [9,1024]
    const float* b_out = (const float*)d_in[6];   // [9]
    float*       out   = (float*)d_out;           // [9]

    char* ws = (char*)d_ws;
    u16*      wpack = (u16*)ws;                                // 8 MB packed bf16 W_hh
    u16*      h_bf  = (u16*)(ws + (size_t)8 * 1024 * 1024);    // 4 KB (2x1024 bf16)
    float*    hfp   = (float*)(ws + (size_t)8 * 1024 * 1024 + 4096);   // 4 KB
    unsigned* bar   = (unsigned*)(ws + (size_t)8 * 1024 * 1024 + 8192);// 8 B

    init_state<<<8, 256, 0, stream>>>(h_bf, bar);
    pack_whh<<<(4096 * 1024) / 256, 256, 0, stream>>>(w_hh, wpack);
    lstm_persistent<<<NBLK, BTHREADS, 0, stream>>>(x, w_ih, b_ih, b_hh,
                                                   wpack, h_bf, hfp, bar);
    head_kernel<<<1, 288, 0, stream>>>(hfp, w_out, b_out, out);
}
